// HaloBlockV2_25709674233977
// MI455X (gfx1250) — compile-verified, matrix operand order checked by reading, not run on hardware
//
#include <hip/hip_runtime.h>

// ---------------------------------------------------------------------------
// Problem constants (from reference)
// ---------------------------------------------------------------------------
#define BATCH   4
#define CH      512
#define HH      64
#define WW      64
#define HWD     (HH * WW)          // 4096
#define PIX     (BATCH * HWD)      // 16384 pixels
#define HEADS   16
#define HD      32                 // head dim
#define BS      8
#define HALO    3
#define WIN     14                 // BS + 2*HALO
#define NWIN    196                // WIN*WIN
#define WINP    224                // NWIN padded to 7*32 for K-chunks
#define MLP_HID 2048

typedef _Float16 v16h __attribute__((ext_vector_type(16)));
typedef _Float16 v8h  __attribute__((ext_vector_type(8)));
typedef _Float16 v2h  __attribute__((ext_vector_type(2)));
typedef float    v8f  __attribute__((ext_vector_type(8)));

__device__ __forceinline__ v16h frag_combine(v8h lo, v8h hi) {
  return __builtin_shufflevector(lo, hi, 0,1,2,3,4,5,6,7,8,9,10,11,12,13,14,15);
}

__device__ __forceinline__ v8f wmma_f16(v16h a, v16h b, v8f c) {
  // D = A(16x32 f16) x B(32x16 f16) + C(16x16 f32)
  return __builtin_amdgcn_wmma_f32_16x16x32_f16(
      /*neg_a=*/false, a, /*neg_b=*/false, b,
      /*c_mod=*/(short)0, c, /*reuse_a=*/false, /*reuse_b=*/false);
}

__device__ __forceinline__ v16h load_frag(const _Float16* p) {
  // p points at row/col start + khalf*8; two contiguous 16B halves (K and K+16)
  v8h lo = *(const v8h*)(p);
  v8h hi = *(const v8h*)(p + 16);
  return frag_combine(lo, hi);
}

// ---------------------------------------------------------------------------
// Weight convert: f32 (K x N) row-major  ->  f16 Bt (N x K) row-major.
// Bt layout makes WMMA B-fragments contiguous 16B loads per lane.
// ---------------------------------------------------------------------------
__global__ __launch_bounds__(256) void wt_convert_kernel(
    const float* __restrict__ w, _Float16* __restrict__ bt, int K, int N) {
  int idx = blockIdx.x * 256 + threadIdx.x;
  if (idx < K * N) {
    int k = idx / N;
    int n = idx - k * N;
    bt[(size_t)n * K + k] = (_Float16)w[idx];
  }
}

// ---------------------------------------------------------------------------
// LayerNorm over C per pixel.  x: (B,C,H,W) f32 -> out: (P,C) f16.
// One wave (32 lanes) per pixel, 8 pixels per 256-thread workgroup.
// ---------------------------------------------------------------------------
__global__ __launch_bounds__(256) void ln1_kernel(
    const float* __restrict__ x, const float* __restrict__ g,
    const float* __restrict__ bparm, _Float16* __restrict__ out) {
  int wave = threadIdx.x >> 5, lane = threadIdx.x & 31;
  int p  = blockIdx.x * 8 + wave;
  int b  = p >> 12;          // / HWD
  int hw = p & 4095;
  const float* xb = x + (size_t)b * CH * HWD + hw;
  float vals[16];
  float s = 0.f, s2 = 0.f;
#pragma unroll
  for (int i = 0; i < 16; ++i) {
    int ch = lane + i * 32;
    float v = xb[(size_t)ch * HWD];
    vals[i] = v; s += v; s2 += v * v;
  }
#pragma unroll
  for (int off = 16; off >= 1; off >>= 1) {
    s  += __shfl_xor(s,  off, 32);
    s2 += __shfl_xor(s2, off, 32);
  }
  float mu  = s * (1.f / CH);
  float var = s2 * (1.f / CH) - mu * mu;
  float inv = rsqrtf(var + 1e-5f);
#pragma unroll
  for (int i = 0; i < 16; ++i) {
    int ch = lane + i * 32;
    out[(size_t)p * CH + ch] = (_Float16)((vals[i] - mu) * inv * g[ch] + bparm[ch]);
  }
}

// ---------------------------------------------------------------------------
// Residual-1 + LayerNorm-2:
//   xr = x + gamma1*attn     (stored f32, NCHW, kept for final residual)
//   y  = LN(xr)*g2 + b2      (stored f16, (P,C), GEMM input)
// ---------------------------------------------------------------------------
__global__ __launch_bounds__(256) void res_ln2_kernel(
    const float* __restrict__ x, const float* __restrict__ attn,
    const float* __restrict__ gamma1, const float* __restrict__ g,
    const float* __restrict__ bparm, float* __restrict__ xr,
    _Float16* __restrict__ yout) {
  int wave = threadIdx.x >> 5, lane = threadIdx.x & 31;
  int p  = blockIdx.x * 8 + wave;
  int b  = p >> 12;
  int hw = p & 4095;
  size_t base = (size_t)b * CH * HWD + hw;
  float vals[16];
  float s = 0.f, s2 = 0.f;
#pragma unroll
  for (int i = 0; i < 16; ++i) {
    int ch = lane + i * 32;
    size_t o = base + (size_t)ch * HWD;
    float v = x[o] + gamma1[ch] * attn[o];
    xr[o] = v;
    vals[i] = v; s += v; s2 += v * v;
  }
#pragma unroll
  for (int off = 16; off >= 1; off >>= 1) {
    s  += __shfl_xor(s,  off, 32);
    s2 += __shfl_xor(s2, off, 32);
  }
  float mu  = s * (1.f / CH);
  float var = s2 * (1.f / CH) - mu * mu;
  float inv = rsqrtf(var + 1e-5f);
#pragma unroll
  for (int i = 0; i < 16; ++i) {
    int ch = lane + i * 32;
    yout[(size_t)p * CH + ch] = (_Float16)((vals[i] - mu) * inv * g[ch] + bparm[ch]);
  }
}

// ---------------------------------------------------------------------------
// WMMA GEMM:  C(MxN) = A(MxK f16, row-major) @ Bt(NxK f16, row-major)^T
// K, N, MODE are compile-time: all fragment addresses fold into the 24-bit
// immediate offset of global_load_b128 (no per-iteration VALU address math).
// 256 threads = 8 waves (2 M x 4 N); each wave owns a 64x32 output tile
// (8 accumulators): per 32-wide k-chunk -> 4 A-frags + 2 B-frags, 8 WMMAs.
// Workgroup tile = 128x128.
// MODE 0: store f16                              (q / kv projections)
// MODE 1: +bias, exact GELU, store f16           (MLP fc1)
// MODE 2: +bias, out = resid + gscale*v, f32 scattered to NCHW (MLP fc2)
// ---------------------------------------------------------------------------
template <int K, int N, int MODE>
__global__ __launch_bounds__(256) void gemm_wmma_kernel(
    const _Float16* __restrict__ A, const _Float16* __restrict__ Bt,
    const float* __restrict__ bias, _Float16* __restrict__ outH,
    float* __restrict__ outF, const float* __restrict__ resid,
    const float* __restrict__ gscale) {
  int tid  = threadIdx.x;
  int wave = tid >> 5, lane = tid & 31;
  int wm = wave >> 2;                    // 0..1
  int wn = wave & 3;                     // 0..3
  int m0 = blockIdx.y * 128 + wm * 64;
  int n0 = blockIdx.x * 128 + wn * 32;
  int khalf = lane >> 4;                 // 16-bit A-layout: lanes>=16 hold K+8
  int r16   = lane & 15;
  const _Float16* Ap = A  + (size_t)(m0 + r16) * K + khalf * 8;
  const _Float16* Bp = Bt + (size_t)(n0 + r16) * K + khalf * 8;
  v8f acc[8];
#pragma unroll
  for (int i = 0; i < 8; ++i) acc[i] = (v8f){0.f,0.f,0.f,0.f,0.f,0.f,0.f,0.f};

#pragma unroll 2
  for (int kc = 0; kc < K; kc += 32) {
    v16h af[4], bf[2];
#pragma unroll
    for (int i = 0; i < 4; ++i) af[i] = load_frag(Ap + i * (16 * K) + kc);
#pragma unroll
    for (int j = 0; j < 2; ++j) bf[j] = load_frag(Bp + j * (16 * K) + kc);
#pragma unroll
    for (int i = 0; i < 4; ++i)
#pragma unroll
      for (int j = 0; j < 2; ++j)
        acc[i * 2 + j] = wmma_f16(af[i], bf[j], acc[i * 2 + j]);
  }

#pragma unroll
  for (int i = 0; i < 4; ++i) {
    int mbase = m0 + i * 16 + (khalf << 3);  // D layout: VGPR r -> M=r (+8 hi)
#pragma unroll
    for (int j = 0; j < 2; ++j) {
      v8f a = acc[i * 2 + j];
      int col = n0 + j * 16 + r16;
      if constexpr (MODE == 0) {
#pragma unroll
        for (int r = 0; r < 8; ++r)
          outH[(size_t)(mbase + r) * N + col] = (_Float16)a[r];
      } else if constexpr (MODE == 1) {
        float bv = bias[col];
#pragma unroll
        for (int r = 0; r < 8; ++r) {
          float v = a[r] + bv;
          v = 0.5f * v * (1.f + erff(v * 0.70710678118f));   // exact GELU
          outH[(size_t)(mbase + r) * N + col] = (_Float16)v;
        }
      } else {
        float bv  = bias[col];
        float gsc = gscale[col];
#pragma unroll
        for (int r = 0; r < 8; ++r) {
          float v = a[r] + bv;
          int m  = mbase + r;
          int bb = m >> 12;
          int hw = m & 4095;
          size_t o = ((size_t)bb * CH + col) * HWD + hw;     // NCHW scatter
          outF[o] = resid[o] + gsc * v;
        }
      }
    }
  }
}

// ---------------------------------------------------------------------------
// Halo attention: one workgroup per (batch, head, 8x8 block).
// Q: 64x32, K/V window: 196x32 padded to 224.  All matmuls via WMMA.
// ---------------------------------------------------------------------------
__global__ __launch_bounds__(256) void halo_attn_kernel(
    const _Float16* __restrict__ q, const _Float16* __restrict__ kv,
    float* __restrict__ attn) {
  __shared__ _Float16 Qs[64 * 32];      //  4 KB
  __shared__ _Float16 Ks[WINP * 32];    // 14 KB  [pos][d]
  __shared__ _Float16 Vt[32 * WINP];    // 14 KB  [d][pos]
  __shared__ _Float16 Sh[64 * WINP];    // 28 KB  scores, then probs (f16)

  int tid  = threadIdx.x;
  int blk  = blockIdx.x;
  int b    = blk >> 10;                 // /(HEADS*64)
  int rem  = blk & 1023;
  int head = rem >> 6;
  int cell = rem & 63;
  int bh0  = cell >> 3, bw0 = cell & 7;

  // ---- gather K/V window (zero-pad halo + K-padding) and Q -------------
  for (int idx = tid; idx < WINP * 16; idx += 256) {
    int pos = idx >> 4;
    int dp  = (idx & 15) << 1;
    _Float16 k0 = (_Float16)0.f, k1 = (_Float16)0.f;
    _Float16 v0 = (_Float16)0.f, v1 = (_Float16)0.f;
    if (pos < NWIN) {
      int wr = pos / WIN, wc = pos - wr * WIN;
      int gr = bh0 * BS - HALO + wr;
      int gc = bw0 * BS - HALO + wc;
      if ((unsigned)gr < (unsigned)HH && (unsigned)gc < (unsigned)WW) {
        size_t base = (((size_t)b * HH + gr) * WW + gc) * (2 * CH) + head * HD + dp;
        v2h kk = *(const v2h*)(kv + base);
        v2h vv = *(const v2h*)(kv + base + CH);
        k0 = kk[0]; k1 = kk[1]; v0 = vv[0]; v1 = vv[1];
      }
    }
    Ks[pos * 32 + dp]     = k0;
    Ks[pos * 32 + dp + 1] = k1;
    Vt[dp * WINP + pos]       = v0;
    Vt[(dp + 1) * WINP + pos] = v1;
  }
  for (int idx = tid; idx < 64 * 16; idx += 256) {
    int m  = idx >> 4;
    int dp = (idx & 15) << 1;
    int gr = bh0 * BS + (m >> 3);
    int gc = bw0 * BS + (m & 7);
    size_t base = (((size_t)b * HH + gr) * WW + gc) * CH + head * HD + dp;
    v2h qq = *(const v2h*)(q + base);
    Qs[m * 32 + dp]     = qq[0];
    Qs[m * 32 + dp + 1] = qq[1];
  }
  __syncthreads();

  int wave  = tid >> 5, lane = tid & 31;
  int khalf = lane >> 4;
  int r16   = lane & 15;

  // ---- S = (Q K^T) * scale: wave owns M-tile (wave&3) and 7 N-tiles ----
  // A fragment loaded once per wave, reused for 7 WMMAs.
  const float scale = 0.17677669529f;   // 1/sqrt(32)
  {
    int mt = wave & 3;
    int ntbase = (wave >> 2) * 7;
    int arow = mt * 16 + r16;
    v16h afrag = load_frag(&Qs[arow * 32 + khalf * 8]);
    int mbase = mt * 16 + (khalf << 3);
    for (int nn = 0; nn < 7; ++nn) {
      int nt = ntbase + nn;
      int bcol = nt * 16 + r16;
      v16h bfrag = load_frag(&Ks[bcol * 32 + khalf * 8]);
      v8f acc = {0.f, 0.f, 0.f, 0.f, 0.f, 0.f, 0.f, 0.f};
      acc = wmma_f16(afrag, bfrag, acc);
      int scol = nt * 16 + r16;
#pragma unroll
      for (int r = 0; r < 8; ++r)
        Sh[(mbase + r) * WINP + scol] = (_Float16)(acc[r] * scale);
    }
  }
  __syncthreads();

  // ---- softmax over 196 valid cols; each wave owns 8 rows exclusively --
  for (int rr = 0; rr < 8; ++rr) {
    int rowi = wave * 8 + rr;
    float vals[7];
    float mx = -1e30f;
#pragma unroll
    for (int i = 0; i < 7; ++i) {
      int c = lane + i * 32;
      float v = (c < NWIN) ? (float)Sh[rowi * WINP + c] : -1e30f;
      vals[i] = v;
      mx = fmaxf(mx, v);
    }
#pragma unroll
    for (int off = 16; off >= 1; off >>= 1) mx = fmaxf(mx, __shfl_xor(mx, off, 32));
    float sum = 0.f;
#pragma unroll
    for (int i = 0; i < 7; ++i) {
      int c = lane + i * 32;
      float e = (c < NWIN) ? __expf(vals[i] - mx) : 0.f;
      vals[i] = e;
      sum += e;
    }
#pragma unroll
    for (int off = 16; off >= 1; off >>= 1) sum += __shfl_xor(sum, off, 32);
    float inv = 1.f / sum;
#pragma unroll
    for (int i = 0; i < 7; ++i) {
      int c = lane + i * 32;
      Sh[rowi * WINP + c] = (_Float16)(vals[i] * inv);   // padded cols -> 0
    }
  }
  __syncthreads();

  // ---- O = P @ V : 4 M-tiles x 2 N-tiles, one per wave; 7 K-chunks -----
  {
    int mt = wave >> 1, nt = wave & 1;
    int arow = mt * 16 + r16;
    int bcol = nt * 16 + r16;             // = output head-dim index
    v8f acc = {0.f, 0.f, 0.f, 0.f, 0.f, 0.f, 0.f, 0.f};
#pragma unroll
    for (int kc = 0; kc < WINP; kc += 32) {
      v16h afrag = load_frag(&Sh[arow * WINP + kc + khalf * 8]);
      v16h bfrag = load_frag(&Vt[bcol * WINP + kc + khalf * 8]);
      acc = wmma_f16(afrag, bfrag, acc);
    }
    int ch = head * HD + nt * 16 + r16;
    int mbase = mt * 16 + (khalf << 3);
#pragma unroll
    for (int r = 0; r < 8; ++r) {
      int m  = mbase + r;
      int gr = bh0 * BS + (m >> 3);
      int gc = bw0 * BS + (m & 7);
      attn[(((size_t)b * CH + ch) * HH + gr) * WW + gc] = acc[r];
    }
  }
}

// ---------------------------------------------------------------------------
// Host-side orchestration
// ---------------------------------------------------------------------------
extern "C" void kernel_launch(void* const* d_in, const int* in_sizes, int n_in,
                              void* d_out, int out_size, void* d_ws, size_t ws_size,
                              hipStream_t stream) {
  (void)in_sizes; (void)n_in; (void)out_size; (void)ws_size;
  const float* x      = (const float*)d_in[0];
  const float* g1     = (const float*)d_in[1];
  const float* b1     = (const float*)d_in[2];
  const float* w_q    = (const float*)d_in[3];
  const float* w_kv   = (const float*)d_in[4];
  const float* gamma1 = (const float*)d_in[5];
  const float* g2     = (const float*)d_in[6];
  const float* b2     = (const float*)d_in[7];
  const float* w_fc1  = (const float*)d_in[8];
  const float* b_fc1  = (const float*)d_in[9];
  const float* w_fc2  = (const float*)d_in[10];
  const float* b_fc2  = (const float*)d_in[11];
  const float* gamma2 = (const float*)d_in[12];
  float* out = (float*)d_out;

  char* ws = (char*)d_ws;
  const size_t MB = 1024ull * 1024ull;
  // Workspace layout (with reuse):
  //   [0,16MB)    xn f16 (P,C)      -> later reused as y (LN2 output)
  //   [16,32MB)   q  f16 (P,C)
  //   [32,64MB)   kv f16 (P,2C)
  //   [64,96MB)   attn f32 NCHW
  //   [16,80MB)   h1 f16 (P,4C)     overlays dead q/kv/attn
  //   [96,128MB)  xr f32 NCHW
  //   [128MB..)   f16 transposed weights
  _Float16* xn   = (_Float16*)(ws + 0);
  _Float16* qbuf = (_Float16*)(ws + 16 * MB);
  _Float16* kvb  = (_Float16*)(ws + 32 * MB);
  float*    attn = (float*)   (ws + 64 * MB);
  _Float16* h1   = (_Float16*)(ws + 16 * MB);
  float*    xr   = (float*)   (ws + 96 * MB);
  _Float16* y    = xn;
  _Float16* wqT  = (_Float16*)(ws + 128 * MB);   // 512x512   (0.5 MB)
  _Float16* wkvT = (_Float16*)(ws + 129 * MB);   // 1024x512  (1 MB)
  _Float16* wf1T = (_Float16*)(ws + 131 * MB);   // 2048x512  (2 MB)
  _Float16* wf2T = (_Float16*)(ws + 134 * MB);   // 512x2048  (2 MB)

  // 1) weights -> f16, transposed to (N,K)
  wt_convert_kernel<<<(CH * CH + 255) / 256, 256, 0, stream>>>(w_q, wqT, CH, CH);
  wt_convert_kernel<<<(CH * 2 * CH + 255) / 256, 256, 0, stream>>>(w_kv, wkvT, CH, 2 * CH);
  wt_convert_kernel<<<(CH * MLP_HID + 255) / 256, 256, 0, stream>>>(w_fc1, wf1T, CH, MLP_HID);
  wt_convert_kernel<<<(MLP_HID * CH + 255) / 256, 256, 0, stream>>>(w_fc2, wf2T, MLP_HID, CH);

  // 2) LN1 -> xn f16 (P,C)
  ln1_kernel<<<PIX / 8, 256, 0, stream>>>(x, g1, b1, xn);

  // 3) q = xn @ w_q    (M=16384, N=512, K=512)
  gemm_wmma_kernel<CH, CH, 0><<<dim3(CH / 128, PIX / 128), 256, 0, stream>>>(
      xn, wqT, nullptr, qbuf, nullptr, nullptr, nullptr);
  // 4) kv = xn @ w_kv  (N=1024)
  gemm_wmma_kernel<CH, 2 * CH, 0><<<dim3(2 * CH / 128, PIX / 128), 256, 0, stream>>>(
      xn, wkvT, nullptr, kvb, nullptr, nullptr, nullptr);

  // 5) halo attention -> attn f32 NCHW
  halo_attn_kernel<<<BATCH * HEADS * 64, 256, 0, stream>>>(qbuf, kvb, attn);

  // 6) xr = x + gamma1*attn ; y = LN2(xr)
  res_ln2_kernel<<<PIX / 8, 256, 0, stream>>>(x, attn, gamma1, g2, b2, xr, y);

  // 7) h1 = gelu(y @ w_fc1 + b_fc1)   (N=2048, K=512)
  gemm_wmma_kernel<CH, MLP_HID, 1><<<dim3(MLP_HID / 128, PIX / 128), 256, 0, stream>>>(
      y, wf1T, b_fc1, h1, nullptr, nullptr, nullptr);

  // 8) out = xr + gamma2 * (h1 @ w_fc2 + b_fc2)   (N=512, K=2048), NCHW f32
  gemm_wmma_kernel<MLP_HID, CH, 2><<<dim3(CH / 128, PIX / 128), 256, 0, stream>>>(
      h1, wf2T, b_fc2, nullptr, out, xr, gamma2);
}
